// SememeEmbeddings_20280835571792
// MI455X (gfx1250) — matrix operand order repaired and checked by reading, not compiled, not run on hardware
//
#include <hip/hip_runtime.h>
#include <hip/hip_bf16.h>

// ---------------------------------------------------------------------------
// SememeEmbeddings on MI455X (gfx1250, wave32, WMMA)
//   out[m, :] = (sqrt(512) / (nnz(row_m)+1e-6)) * (word2sememe[x[m]] @ W)
// Strategy: gather+convert A rows (binary -> exact in bf16) into LDS in WMMA
// A-fragment layout; W pre-split into bf16 hi/lo (error-free compensation)
// and pre-swizzled into WMMA B-fragment layout in workspace (L2 resident).
// ---------------------------------------------------------------------------

typedef __attribute__((ext_vector_type(16))) __bf16 v16bf;
typedef __attribute__((ext_vector_type(8)))  __bf16 v8bf;
typedef __attribute__((ext_vector_type(8)))  float  v8f;

#define K_SEM    2048
#define D_MODEL  512
#define NTOK     16384      // B * S = 8 * 2048
#define M_TILE   32         // rows per block (2 WMMA M-stripes)
#define SCALE_F  22.62741699796952f  // sqrt(512)

// ---------------------------------------------------------------------------
// Prep: W[f32, K x N] -> Wp_hi/Wp_lo[bf16] in B-fragment order.
// Fragment addressing used by the GEMM:
//   lane half = lane>>4, col n = 16*ntile + (lane&15)
//   lane holds K = kt*32 + half*16 + j (j=0..15), contiguous 32B.
//   element index = ((kt*2 + half)*512 + n)*16 + j
// One thread per 16-element run: reads are row-coalesced across the wave,
// writes are one contiguous 32B store per thread.
// ---------------------------------------------------------------------------
__global__ __launch_bounds__(256) void sememe_prep_w(
    const float* __restrict__ W, __bf16* __restrict__ wp_hi,
    __bf16* __restrict__ wp_lo) {
  const int tid = blockIdx.x * 256 + threadIdx.x;  // 0 .. 64*2*512-1
  const int n   = tid & 511;
  const int kh  = tid >> 9;                        // kt*2 + half  (k base = kh*16)
  const float* src = W + (size_t)(kh * 16) * D_MODEL + n;
  v16bf vh, vl;
#pragma unroll
  for (int j = 0; j < 16; ++j) {
    const float f = src[(size_t)j * D_MODEL];
    const __bf16 h = (__bf16)f;
    vh[j] = h;
    vl[j] = (__bf16)(f - (float)h);
  }
  *(v16bf*)(wp_hi + (size_t)tid * 16) = vh;
  *(v16bf*)(wp_lo + (size_t)tid * 16) = vl;
}

// ---------------------------------------------------------------------------
// Main kernel: 512 blocks x 256 threads (8 waves).
//   block: 32 token rows; wave w: N columns [w*64, w*64+64) (4 N-tiles).
// ---------------------------------------------------------------------------
__global__ __launch_bounds__(256) void sememe_gemm(
    const int* __restrict__ x, const float* __restrict__ w2s,
    const __bf16* __restrict__ wp_hi, const __bf16* __restrict__ wp_lo,
    float* __restrict__ out) {
  // A fragments for one K-chunk of 512: [kt_local(16)][stripe(2)][lane(32)][16 bf16]
  __shared__ __align__(16) __bf16 a_lds[16 * 2 * 32 * 16];  // 32 KB
  __shared__ float rowsum[M_TILE];

  const int tid   = threadIdx.x;
  const int lane  = tid & 31;
  const int wave  = tid >> 5;
  const int row   = tid >> 3;   // staging: 8 threads per token row (0..31)
  const int kpart = tid & 7;
  const int m0    = blockIdx.x * M_TILE;

  const int   tok = x[m0 + row];
  const float* rp = w2s + (size_t)tok * K_SEM;

  const int half   = lane >> 4;
  const int ncol   = lane & 15;
  const int r16    = row & 15;
  const int st_stg = row >> 4;  // which M-stripe this staging row belongs to

  float psum = 0.0f;            // partial sememe count for `row`
  v8f acc[2][4] = {};           // [stripe][ntile] f32 accumulators

  for (int chunk = 0; chunk < 4; ++chunk) {
    // ---- stage 512 K-values of each of 32 rows: f32 -> bf16 A-fragments ----
    const int kbase = chunk * 512 + kpart * 64;
#pragma unroll
    for (int g = 0; g < 8; ++g) {
      const int k = kbase + g * 8;
      const float4 f0 = *(const float4*)(rp + k);
      const float4 f1 = *(const float4*)(rp + k + 4);
      psum += f0.x + f0.y + f0.z + f0.w + f1.x + f1.y + f1.z + f1.w;
      v8bf v;
      v[0] = (__bf16)f0.x; v[1] = (__bf16)f0.y;
      v[2] = (__bf16)f0.z; v[3] = (__bf16)f0.w;
      v[4] = (__bf16)f1.x; v[5] = (__bf16)f1.y;
      v[6] = (__bf16)f1.z; v[7] = (__bf16)f1.w;
      // A 16x32 layout (matches B): lanes 0-15 hold K{0..15},
      // lanes 16-31 hold K{16..31} of row (lane&15).
      const int ktl   = (k >> 5) & 15;           // kt within chunk
      const int sub   = (k >> 3) & 3;            // which 8-run of the 32
      const int dlane = r16 + ((sub >> 1) << 4);
      const int slot  = (sub & 1) << 3;          // 0 or 8
      *(v8bf*)&a_lds[(((ktl * 2 + st_stg) * 32) + dlane) * 16 + slot] = v;
    }
    __syncthreads();

    // ---- WMMA sweep over this K-chunk ----
#pragma unroll 2
    for (int ktl = 0; ktl < 16; ++ktl) {
      const v16bf a0  = *(const v16bf*)&a_lds[((ktl * 2 + 0) * 32 + lane) * 16];
      const v16bf a1  = *(const v16bf*)&a_lds[((ktl * 2 + 1) * 32 + lane) * 16];
      const int   ktg = chunk * 16 + ktl;
#pragma unroll
      for (int t = 0; t < 4; ++t) {
        const int    n    = (wave * 4 + t) * 16 + ncol;
        const size_t boff = (((size_t)ktg * 2 + half) * 512 + n) * 16;
        const v16bf bh = *(const v16bf*)(wp_hi + boff);
        const v16bf bl = *(const v16bf*)(wp_lo + boff);
        acc[0][t] = __builtin_amdgcn_wmma_f32_16x16x32_bf16(
            false, a0, false, bh, (short)0, acc[0][t], false, false);
        acc[1][t] = __builtin_amdgcn_wmma_f32_16x16x32_bf16(
            false, a1, false, bh, (short)0, acc[1][t], false, false);
        acc[0][t] = __builtin_amdgcn_wmma_f32_16x16x32_bf16(
            false, a0, false, bl, (short)0, acc[0][t], false, false);
        acc[1][t] = __builtin_amdgcn_wmma_f32_16x16x32_bf16(
            false, a1, false, bl, (short)0, acc[1][t], false, false);
      }
    }
    __syncthreads();
  }

  // ---- per-row sememe counts: reduce 8 staging partials (wave32 shuffles) ----
  psum += __shfl_xor(psum, 4, 8);
  psum += __shfl_xor(psum, 2, 8);
  psum += __shfl_xor(psum, 1, 8);
  if (kpart == 0) rowsum[row] = psum;
  __syncthreads();

  // ---- scale + store. C/D layout: VGPR v, lane -> M = v + 8*(lane>=16), N = lane&15
  const int mh = half * 8;
#pragma unroll
  for (int s = 0; s < 2; ++s) {
    float sc[8];
#pragma unroll
    for (int v = 0; v < 8; ++v)
      sc[v] = SCALE_F / (rowsum[s * 16 + mh + v] + 1e-6f);
#pragma unroll
    for (int t = 0; t < 4; ++t) {
      const int n = (wave * 4 + t) * 16 + ncol;
#pragma unroll
      for (int v = 0; v < 8; ++v)
        out[(size_t)(m0 + s * 16 + mh + v) * D_MODEL + n] = acc[s][t][v] * sc[v];
    }
  }
}

// ---------------------------------------------------------------------------
extern "C" void kernel_launch(void* const* d_in, const int* in_sizes, int n_in,
                              void* d_out, int out_size, void* d_ws,
                              size_t ws_size, hipStream_t stream) {
  const int*   x   = (const int*)d_in[0];     // [8, 2048] token ids
  const float* w2s = (const float*)d_in[1];   // [32000, 2048]
  const float* W   = (const float*)d_in[2];   // [2048, 512]
  float*       out = (float*)d_out;           // [8, 2048, 512]

  __bf16* wp_hi = (__bf16*)d_ws;                          // 2 MB
  __bf16* wp_lo = wp_hi + (size_t)K_SEM * D_MODEL;        // +2 MB

  // 64 kt * 2 halves * 512 cols = 65536 runs of 16 elements
  sememe_prep_w<<<(K_SEM / 16) * D_MODEL / 256, 256, 0, stream>>>(W, wp_hi, wp_lo);
  sememe_gemm<<<NTOK / M_TILE, 256, 0, stream>>>(x, w2s, wp_hi, wp_lo, out);
}
